// attention_model_10239202033772
// MI455X (gfx1250) — matrix operand order repaired and checked by reading, not compiled, not run on hardware
//
#include <hip/hip_runtime.h>
#include <hip/hip_bf16.h>
#include <math.h>

// ---------------- problem constants ----------------
#define TT       16
#define BB       32
#define CIN      2048
#define HH       7
#define WW       7
#define PP       49          // H*W
#define MEMC     512
#define CM       2560        // CIN + MEM
#define CS       2049        // CIN + 1
#define CCAM     1000
#define NCLS     8

#define NPAD     64          // padded positions per image (49 -> 64)
#define NP       (BB * NPAD) // 2048 GEMM N
#define KT32     720         // (CM*9)/32 k-tiles
#define NSTAGE   360         // 2 k-tiles (K=64) per LDS stage
#define STG_US   8192        // ushorts per stage chunk: 2*8*32*16 (16 KB)

typedef __attribute__((ext_vector_type(16))) __bf16 v16bf;
typedef __attribute__((ext_vector_type(8)))  float  v8f;

__device__ __forceinline__ unsigned short f2bf(float f) {
    unsigned int u = __float_as_uint(f);
    unsigned int r = u + 0x7FFFu + ((u >> 16) & 1u);   // round-to-nearest-even
    return (unsigned short)(r >> 16);
}
__device__ __forceinline__ float sigm(float x) { return 1.0f / (1.0f + __expf(-x)); }

// ---------------- generic zero fill ----------------
__global__ void zero_u32(unsigned int* p, long long n) {
    long long i = (long long)blockIdx.x * 256 + threadIdx.x;
    if (i < n) p[i] = 0u;
}

// ---------------- weight pack: fp32 [M][CM][3][3] -> bf16 WMMA A-lane layout ----------------
// layout: [gate][by(4)][kt(720)][mtl(8)][lane(32)][16]  (per (gate,by): contiguous 5.9 MB stream)
__global__ void pack_weights(const float* __restrict__ Wfi, const float* __restrict__ Wc,
                             const float* __restrict__ Wo, unsigned short* __restrict__ Apack) {
    long long id = (long long)blockIdx.x * 256 + threadIdx.x;
    if (id >= 3LL * 4 * KT32 * 8 * 32) return;
    int lane = (int)(id & 31);
    int mtl  = (int)((id >> 5) & 7);
    int kt   = (int)((id >> 8) % KT32);
    int gby  = (int)(id / (256LL * KT32));
    int by   = gby & 3;
    int g    = gby >> 2;
    const float* W = (g == 0) ? Wfi : ((g == 1) ? Wc : Wo);
    int m    = (by * 8 + mtl) * 16 + (lane & 15);
    int tap  = kt / 80;                 // CM/32 = 80 k-tiles per tap
    int cb   = (kt % 80) * 32;
    unsigned short* dst = Apack + id * 16;
#pragma unroll
    for (int i = 0; i < 16; ++i) {
        // A k order used here (contiguous, B matches): lanes<16 hold K={0..15}, lanes>=16 hold K={16..31}
        int koff = i + ((lane >> 4) << 4);
        int c = cb + koff;
        dst[i] = f2bf(W[((size_t)m * CM + c) * 9 + tap]);
    }
}

// ---------------- per-timestep CAM + attention branch ----------------
__global__ __launch_bounds__(256) void attn_step(
    const float* __restrict__ x,      // [B][CIN][49] (this t)
    const float* __restrict__ Wcam,   // [1000][2048]
    const float* __restrict__ Wsfi, const float* __restrict__ bsfi,
    const float* __restrict__ Wsc,  const float* __restrict__ bsc,
    const float* __restrict__ Wso,  const float* __restrict__ bso,
    float* __restrict__ a_st, float* __restrict__ cs_st,     // [B][49]
    unsigned short* __restrict__ comb)                       // [B][9][9][CM] bf16
{
    const int b = blockIdx.x, tid = threadIdx.x;
    __shared__ float pooled[CIN];
    __shared__ float redv[256];
    __shared__ int   redi[256];
    __shared__ float camv[PP];
    __shared__ float attmap[PP];
    __shared__ float gfi[PP], gc[PP], go[PP];
    __shared__ float av[PP], asm_[PP];
    __shared__ int   s_idx;
    const float* xb = x + (size_t)b * CIN * PP;

    // 1) channel-mean pool
    for (int c = tid; c < CIN; c += 256) {
        float s = 0.f;
        for (int p = 0; p < PP; ++p) s += xb[c * PP + p];
        pooled[c] = s * (1.f / 49.f);
    }
    __syncthreads();

    // 2) CAM logits + argmax (first-max tie break)
    float best = -3.0e38f; int bi = CCAM;
    for (int j = tid; j < CCAM; j += 256) {
        const float* wr = Wcam + (size_t)j * CIN;
        float d = 0.f;
        for (int c = 0; c < CIN; ++c) d += pooled[c] * wr[c];
        if (d > best) { best = d; bi = j; }
    }
    redv[tid] = best; redi[tid] = bi;
    __syncthreads();
    for (int s = 128; s > 0; s >>= 1) {
        if (tid < s) {
            if (redv[tid + s] > redv[tid] ||
                (redv[tid + s] == redv[tid] && redi[tid + s] < redi[tid])) {
                redv[tid] = redv[tid + s]; redi[tid] = redi[tid + s];
            }
        }
        __syncthreads();
    }
    if (tid == 0) s_idx = redi[0];
    __syncthreads();

    // 3) CAM map + spatial softmax
    if (tid < PP) {
        const float* wsel = Wcam + (size_t)s_idx * CIN;
        float s = 0.f;
        for (int c = 0; c < CIN; ++c) s += wsel[c] * xb[c * PP + tid];
        camv[tid] = s;
    }
    __syncthreads();
    if (tid == 0) {
        float mx = camv[0];
        for (int p = 1; p < PP; ++p) mx = fmaxf(mx, camv[p]);
        float sum = 0.f;
        for (int p = 0; p < PP; ++p) { float e = __expf(camv[p] - mx); attmap[p] = e; sum += e; }
        float inv = 1.f / sum;
        for (int p = 0; p < PP; ++p) attmap[p] *= inv;
    }
    __syncthreads();

    // 4) attention-branch 3x3 convs (1 out channel each), threads 0..48
    if (tid < PP) {
        int h = tid / 7, w = tid % 7;
        float afi = 0.f, ac = 0.f, ao = 0.f;
        for (int c = 0; c < CS; ++c) {
            const float* row = (c < CIN) ? (xb + (size_t)c * PP) : (a_st + b * PP);
            const float* wf = Wsfi + c * 9;
            const float* wc = Wsc  + c * 9;
            const float* wo = Wso  + c * 9;
#pragma unroll
            for (int ky = 0; ky < 3; ++ky) {
                int hy = h + ky - 1; if (hy < 0 || hy > 6) continue;
#pragma unroll
                for (int kx = 0; kx < 3; ++kx) {
                    int wx = w + kx - 1; if (wx < 0 || wx > 6) continue;
                    float v = row[hy * 7 + wx];
                    int tp = ky * 3 + kx;
                    afi += wf[tp] * v; ac += wc[tp] * v; ao += wo[tp] * v;
                }
            }
        }
        gfi[tid] = afi + bsfi[0]; gc[tid] = ac + bsc[0]; go[tid] = ao + bso[0];
    }
    __syncthreads();

    // 5) attention state update + softmax over a
    if (tid < PP) {
        float fi = sigm(gfi[tid]);
        float o  = sigm(go[tid]);
        float cs = fi * cs_st[b * PP + tid] + (1.f - fi) * tanhf(gc[tid]);
        float a  = o * tanhf(cs) * attmap[tid];
        cs_st[b * PP + tid] = cs;
        a_st[b * PP + tid]  = a;
        av[tid] = a;
    }
    __syncthreads();
    if (tid == 0) {
        float mx = av[0];
        for (int p = 1; p < PP; ++p) mx = fmaxf(mx, av[p]);
        float sum = 0.f;
        for (int p = 0; p < PP; ++p) { float e = __expf(av[p] - mx); asm_[p] = e; sum += e; }
        float inv = 1.f / sum;
        for (int p = 0; p < PP; ++p) asm_[p] *= inv;
    }
    __syncthreads();

    // 6) attended input -> bf16 into padded comb (channels [0, CIN))
    for (int i = tid; i < CIN * PP; i += 256) {
        int c = i / PP, p = i % PP;
        float v = xb[c * PP + p] * asm_[p];
        comb[(size_t)(b * 81 + (p / 7 + 1) * 9 + (p % 7 + 1)) * CM + c] = f2bf(v);
    }
}

// ---------------- main gate GEMM: 9-tap shifted implicit conv via bf16 WMMA ----------------
// grid (N/128=16, M/128=4, gates=3), block 256 (8 waves: 2 M-groups x 4 N-groups)
// A (weights) double-buffered through LDS via async global->LDS copies (ASYNCcnt),
// B (activations) per-lane direct 32B loads. Each wave: 4 Mtiles x 2 Ntiles, 16 WMMA/stage.
__global__ __launch_bounds__(256) void gemm_gates(
    const unsigned short* __restrict__ Apack,   // [3][4][720][8][32][16] bf16
    const unsigned short* __restrict__ comb,    // [B][9][9][CM] bf16
    float* __restrict__ gpre)                   // [3][512][2048]
{
    __shared__ unsigned short Abuf[2][STG_US];  // 2 x 16 KB double buffer

    const int tid  = threadIdx.x;
    const int lane = tid & 31;
    const int wv   = tid >> 5;
    const int wv_m = wv & 1;
    const int wv_n = wv >> 1;
    const int gate = blockIdx.z;
    const int nt0  = blockIdx.x * 8 + wv_n * 2;
    const int half = (lane >> 4) << 4;          // K-half channels for B tile: 0 or 16

    // per-lane N-column info (B-matrix: lane -> column N = lane%16)
    int nb[2], nh[2], nw[2];
#pragma unroll
    for (int nt = 0; nt < 2; ++nt) {
        int n = (nt0 + nt) * 16 + (lane & 15);
        int b = n >> 6, p = n & 63;
        if (p > 48) p = 48;                      // padded columns: duplicate, discarded on store
        nb[nt] = b; nh[nt] = p / 7; nw[nt] = p % 7;
    }

    v8f acc[4][2];
#pragma unroll
    for (int mi = 0; mi < 4; ++mi)
#pragma unroll
        for (int nt = 0; nt < 2; ++nt) acc[mi][nt] = (v8f)0.0f;

    // this block's contiguous A stream (5.9 MB over 360 stages of 16 KB)
    const unsigned short* Ablk =
        Apack + (size_t)(gate * 4 + blockIdx.y) * KT32 * 8 * 32 * 16;
    const unsigned lds0 = (unsigned)(uintptr_t)(&Abuf[0][0]);
    const unsigned lds1 = (unsigned)(uintptr_t)(&Abuf[1][0]);

    // async copy of one 16 KB stage into LDS buffer (4 x b128 per thread)
    auto issueA = [&](int s, int buf) {
        unsigned long long src = (unsigned long long)(Ablk + (size_t)s * STG_US) + tid * 16u;
        unsigned ldst = (buf ? lds1 : lds0) + (unsigned)tid * 16u;
#pragma unroll
        for (int i = 0; i < 4; ++i) {
            asm volatile("global_load_async_to_lds_b128 %0, %1, off"
                         :: "v"(ldst + i * 4096u), "v"(src + i * 4096ull)
                         : "memory");
        }
    };

    issueA(0, 0);

    int s = 0;
    for (int tap = 0; tap < 9; ++tap) {
        const int dy = tap / 3, dx = tap % 3;
        const unsigned short* bbase[2];
#pragma unroll
        for (int nt = 0; nt < 2; ++nt)
            bbase[nt] = comb +
                (size_t)(nb[nt] * 81 + (nh[nt] + dy) * 9 + (nw[nt] + dx)) * CM + half;

        for (int ss = 0; ss < 40; ++ss, ++s) {
            asm volatile("s_wait_asynccnt 0x0" ::: "memory");  // own stage-s copies done
            __syncthreads();                                   // everyone's copies visible
            if (s + 1 < NSTAGE) issueA(s + 1, (s + 1) & 1);    // overlap next copy w/ compute
            const unsigned short* ab = &Abuf[s & 1][0];

#pragma unroll
            for (int ktl = 0; ktl < 2; ++ktl) {
                const int kc = ss * 2 + ktl;
                v16bf a[4], bvec[2];
#pragma unroll
                for (int mi = 0; mi < 4; ++mi)
                    a[mi] = *(const v16bf*)(ab +
                        (size_t)((ktl * 8 + wv_m * 4 + mi) * 32 + lane) * 16);
#pragma unroll
                for (int nt = 0; nt < 2; ++nt)
                    bvec[nt] = *(const v16bf*)(bbase[nt] + kc * 32);
#pragma unroll
                for (int mi = 0; mi < 4; ++mi)
#pragma unroll
                    for (int nt = 0; nt < 2; ++nt)
                        acc[mi][nt] = __builtin_amdgcn_wmma_f32_16x16x32_bf16(
                            false, a[mi], false, bvec[nt], (short)0, acc[mi][nt],
                            false, false);
            }
            __syncthreads();                                   // WAR: done reading buffer
        }
    }

    // epilogue: C 16x16 f32 layout: lane L, vgpr v -> M = v + 8*(L/16), N = L%16
    float* gp = gpre + (size_t)gate * MEMC * NP;
    const int mt0 = blockIdx.y * 8 + wv_m * 4;
    const int mhi = (lane >> 4) * 8;
    const int ncl = lane & 15;
#pragma unroll
    for (int mi = 0; mi < 4; ++mi)
#pragma unroll
        for (int nt = 0; nt < 2; ++nt) {
            int mrow0 = (mt0 + mi) * 16 + mhi;
            int ncol  = (nt0 + nt) * 16 + ncl;
#pragma unroll
            for (int v = 0; v < 8; ++v)
                gp[(size_t)(mrow0 + v) * NP + ncol] = acc[mi][nt][v];
        }
}

// ---------------- ConvLSTM state update ----------------
__global__ void state_step(const float* __restrict__ gpre,
                           const float* __restrict__ bfi, const float* __restrict__ bc,
                           const float* __restrict__ bo,
                           float* __restrict__ c_st, float* __restrict__ h_st,
                           unsigned short* __restrict__ comb) {
    int i = blockIdx.x * 256 + threadIdx.x;       // over B*MEM*49
    if (i >= BB * MEMC * PP) return;
    int p = i % PP;
    int m = (i / PP) % MEMC;
    int b = i / (PP * MEMC);
    int n = b * NPAD + p;
    float fi = sigm(gpre[(size_t)(0 * MEMC + m) * NP + n] + bfi[m]);
    float o  = sigm(gpre[(size_t)(2 * MEMC + m) * NP + n] + bo[m]);
    float cc = tanhf(gpre[(size_t)(1 * MEMC + m) * NP + n] + bc[m]);
    float c  = fi * c_st[i] + (1.f - fi) * cc;
    float h  = o * tanhf(c);
    c_st[i] = c; h_st[i] = h;
    comb[(size_t)(b * 81 + (p / 7 + 1) * 9 + (p % 7 + 1)) * CM + CIN + m] = f2bf(h);
}

// ---------------- final head: avgpool + FC ----------------
__global__ __launch_bounds__(256) void head(const float* __restrict__ h_st,
                                            const float* __restrict__ Wfc,
                                            const float* __restrict__ bfc,
                                            float* __restrict__ out) {
    int tid = threadIdx.x;
    for (int i = tid; i < BB * MEMC; i += 256) {
        float s = 0.f;
        for (int p = 0; p < PP; ++p) s += h_st[(size_t)i * PP + p];
        out[BB * NCLS + i] = s * (1.f / 49.f);    // feats after logits
    }
    __syncthreads();
    int b = tid >> 3, j = tid & 7;                // 32*8 == 256
    const float* f = out + BB * NCLS + b * MEMC;
    float s = bfc[j];
    for (int m = 0; m < MEMC; ++m) s += f[m] * Wfc[j * MEMC + m];
    out[b * NCLS + j] = s;
}

// ---------------- host launch ----------------
extern "C" void kernel_launch(void* const* d_in, const int* in_sizes, int n_in,
                              void* d_out, int out_size, void* d_ws, size_t ws_size,
                              hipStream_t stream) {
    const float* feats = (const float*)d_in[0];
    const float* Wfi  = (const float*)d_in[1];
    const float* bfi  = (const float*)d_in[2];
    const float* Wc   = (const float*)d_in[3];
    const float* bc   = (const float*)d_in[4];
    const float* Wo   = (const float*)d_in[5];
    const float* bo   = (const float*)d_in[6];
    const float* Wsfi = (const float*)d_in[7];
    const float* bsfi = (const float*)d_in[8];
    const float* Wsc  = (const float*)d_in[9];
    const float* bsc  = (const float*)d_in[10];
    const float* Wso  = (const float*)d_in[11];
    const float* bso  = (const float*)d_in[12];
    const float* Wcam = (const float*)d_in[13];
    const float* Wfc  = (const float*)d_in[14];
    const float* bfc  = (const float*)d_in[15];
    float* out = (float*)d_out;

    char* ws = (char*)d_ws;
    const size_t SZ_COMB  = (size_t)BB * 81 * CM * 2;                  // 13,271,040
    const size_t SZ_APACK = (size_t)3 * KT32 * 32 * 32 * 16 * 2;       // 70,778,880
    const size_t SZ_GPRE  = (size_t)3 * MEMC * NP * 4;                 // 12,582,912
    const size_t SZ_ST    = (size_t)BB * MEMC * PP * 4;                //  3,211,264
    unsigned short* comb  = (unsigned short*)ws;
    unsigned short* Apack = (unsigned short*)(ws + SZ_COMB);
    float* gpre  = (float*)(ws + SZ_COMB + SZ_APACK);
    float* c_st  = (float*)(ws + SZ_COMB + SZ_APACK + SZ_GPRE);
    float* h_st  = (float*)(ws + SZ_COMB + SZ_APACK + SZ_GPRE + SZ_ST);
    float* a_st  = (float*)(ws + SZ_COMB + SZ_APACK + SZ_GPRE + 2 * SZ_ST);
    float* cs_st = a_st + BB * PP;   // contiguous after a_st

    // per-launch zero init (borders of comb, h-part of comb, c state, attn states)
    {
        long long n0 = (long long)(SZ_COMB / 4);
        zero_u32<<<(unsigned)((n0 + 255) / 256), 256, 0, stream>>>((unsigned int*)comb, n0);
        long long n1 = (long long)BB * MEMC * PP;
        zero_u32<<<(unsigned)((n1 + 255) / 256), 256, 0, stream>>>((unsigned int*)c_st, n1);
        long long n2 = 2LL * BB * PP;
        zero_u32<<<(unsigned)((n2 + 255) / 256), 256, 0, stream>>>((unsigned int*)a_st, n2);
    }

    // pack conv weights to bf16 WMMA layout (amortized over 16 timesteps)
    {
        long long nrec = 3LL * 4 * KT32 * 8 * 32;
        pack_weights<<<(unsigned)((nrec + 255) / 256), 256, 0, stream>>>(Wfi, Wc, Wo, Apack);
    }

    for (int t = 0; t < TT; ++t) {
        const float* xt = feats + (size_t)t * BB * CIN * PP;
        attn_step<<<BB, 256, 0, stream>>>(xt, Wcam, Wsfi, bsfi, Wsc, bsc, Wso, bso,
                                          a_st, cs_st, comb);
        dim3 grid(NP / 128, MEMC / 128, 3);
        gemm_gates<<<grid, 256, 0, stream>>>(Apack, comb, gpre);
        int nelem = BB * MEMC * PP;
        state_step<<<(nelem + 255) / 256, 256, 0, stream>>>(gpre, bfi, bc, bo,
                                                            c_st, h_st, comb);
    }

    head<<<1, 256, 0, stream>>>(h_st, Wfc, bfc, out);
    (void)in_sizes; (void)n_in; (void)out_size; (void)ws_size;
}
